// pygGCN_61065845015370
// MI455X (gfx1250) — matrix operand order repaired and checked by reading, not compiled, not run on hardware
//
#include <hip/hip_runtime.h>
#include <hip/hip_bf16.h>
#include <stdint.h>

// ---------------------------------------------------------------------------
// Types for WMMA fragments (wave32, gfx1250)
// ---------------------------------------------------------------------------
typedef float  v8f   __attribute__((ext_vector_type(8)));
typedef __bf16 v8bf  __attribute__((ext_vector_type(8)));
typedef __bf16 v16bf __attribute__((ext_vector_type(16)));

#define GEMM_BM 128
#define GEMM_BN 64
#define GEMM_BK 32
#define LDS_A   40   // BK + 8 bf16 pad -> 80B row stride, conflict-free b128 reads
#define LDS_B   40

// ---------------------------------------------------------------------------
// Global -> LDS staging: async-to-LDS when available (ASYNCcnt path), else
// plain b128 load + ds_store fallback.
// ---------------------------------------------------------------------------
#if defined(__gfx1250__) && __has_builtin(__builtin_amdgcn_global_load_async_to_lds_b128)
#define HAS_ASYNC_LDS 1
#else
#define HAS_ASYNC_LDS 0
#endif

typedef int v4i __attribute__((vector_size(4 * sizeof(int))));
typedef __attribute__((address_space(1))) v4i gv4i;   // global
typedef __attribute__((address_space(3))) v4i lv4i;   // LDS

__device__ __forceinline__ void stage16(void* lds_dst, const void* gsrc) {
#if HAS_ASYNC_LDS
  __builtin_amdgcn_global_load_async_to_lds_b128(
      (gv4i*)(uintptr_t)gsrc,
      (lv4i*)(uint32_t)(uintptr_t)lds_dst,
      /*offset=*/0, /*cpol=*/0);
#else
  *(uint4*)lds_dst = *(const uint4*)gsrc;
#endif
}

__device__ __forceinline__ void stage_wait() {
#if HAS_ASYNC_LDS
#if __has_builtin(__builtin_amdgcn_s_wait_asynccnt)
  __builtin_amdgcn_s_wait_asynccnt(0);
#else
  asm volatile("s_wait_asynccnt 0" ::: "memory");
#endif
#endif
}

// ---------------------------------------------------------------------------
// GEMM: C[M,N] (f32) = A[M,K] (bf16, row-major) * Bt[N,K]^T (bf16, row-major
// transposed weights).  256 threads = 8 waves; wave w owns rows [16w,16w+16)
// of the 128x64 block tile and 4 accumulators of 16x16.
// ---------------------------------------------------------------------------
__global__ __launch_bounds__(256, 2) void gemm_bf16_wmma(
    const __bf16* __restrict__ A, const __bf16* __restrict__ Bt,
    float* __restrict__ C, int M, int N, int K) {
  __shared__ __bf16 As[2][GEMM_BM * LDS_A];
  __shared__ __bf16 Bs[2][GEMM_BN * LDS_B];

  const int tid  = threadIdx.x;
  const int wave = tid >> 5;
  const int lane = tid & 31;
  const int half = lane >> 4;   // lane group g (0 or 1)
  const int l16  = lane & 15;
  const long long bm = (long long)blockIdx.x * GEMM_BM;
  const long long bn = (long long)blockIdx.y * GEMM_BN;

  v8f acc[4] = {};

  auto stage = [&](int buf, int kt) {
    const long long k0 = (long long)kt * GEMM_BK;
    // A tile: 128 rows x 64B = 512 x 16B chunks, 2 per thread
#pragma unroll
    for (int i = 0; i < 2; ++i) {
      int c    = tid + 256 * i;
      int row  = c >> 2;
      int koff = (c & 3) * 8;                   // bf16 elements
      long long gr = bm + row; if (gr >= M) gr = M - 1;   // clamp (stores guarded)
      stage16(&As[buf][row * LDS_A + koff], A + gr * (long long)K + k0 + koff);
    }
    // B tile: 64 rows x 64B = 256 x 16B chunks, 1 per thread
    {
      int row  = tid >> 2;
      int koff = (tid & 3) * 8;
      stage16(&Bs[buf][row * LDS_B + koff], Bt + (bn + row) * (long long)K + k0 + koff);
    }
  };

  auto compute = [&](int buf) {
    const __bf16* as = As[buf];
    const __bf16* bs = Bs[buf];
    // A fragment (16x32 bf16): lane half g -> elems = K[16g+e]
    const int mrow = (wave << 4) + l16;
    v8bf alo = *(const v8bf*)(as + mrow * LDS_A + 16 * half);
    v8bf ahi = *(const v8bf*)(as + mrow * LDS_A + 16 * half + 8);
    v16bf a  = __builtin_shufflevector(alo, ahi, 0,1,2,3,4,5,6,7,8,9,10,11,12,13,14,15);
#pragma unroll
    for (int nt = 0; nt < 4; ++nt) {
      // B fragment (32x16): lane = column n, half g -> elems = K[16g+e]
      const int nrow = nt * 16 + l16;
      v8bf blo = *(const v8bf*)(bs + nrow * LDS_B + 16 * half);
      v8bf bhi = *(const v8bf*)(bs + nrow * LDS_B + 16 * half + 8);
      v16bf b  = __builtin_shufflevector(blo, bhi, 0,1,2,3,4,5,6,7,8,9,10,11,12,13,14,15);
      acc[nt] = __builtin_amdgcn_wmma_f32_16x16x32_bf16(
          false, a, false, b, (short)0, acc[nt], false, false);
    }
  };

  stage(0, 0);
  stage_wait();
  __syncthreads();
  const int nk = K / GEMM_BK;
  for (int kt = 0; kt < nk; ++kt) {
    const int cur = kt & 1;
    if (kt + 1 < nk) stage(cur ^ 1, kt + 1);
    compute(cur);
    if (kt + 1 < nk) { stage_wait(); __syncthreads(); }
  }

  // D layout: VGPR r holds M = r + 8*(lane/16), N = lane%16
#pragma unroll
  for (int nt = 0; nt < 4; ++nt) {
#pragma unroll
    for (int r = 0; r < 8; ++r) {
      long long m = bm + (wave << 4) + (half << 3) + r;
      if (m < M) C[m * (long long)N + bn + nt * 16 + l16] = acc[nt][r];
    }
  }
}

// ---------------------------------------------------------------------------
// GCN-norm + propagation helpers
// ---------------------------------------------------------------------------
__global__ void init_deg(float* deg, int n) {
  int i = blockIdx.x * blockDim.x + threadIdx.x;
  if (i < n) deg[i] = 1.0f;   // self-loop weight 1
}

__global__ void scatter_deg(const int* __restrict__ ei, const float* __restrict__ ew,
                            float* deg, int E) {
  int e = blockIdx.x * blockDim.x + threadIdx.x;
  if (e < E) atomicAdd(&deg[ei[E + e]], ew[e]);   // col = target
}

__global__ void finalize_dinv(float* d, int n) {
  int i = blockIdx.x * blockDim.x + threadIdx.x;
  if (i < n) { float v = d[i]; d[i] = v > 0.f ? rsqrtf(v) : 0.f; }
}

__global__ void cvt_bf16(const float* __restrict__ s, __bf16* __restrict__ d, long long n) {
  long long i = (long long)blockIdx.x * blockDim.x + threadIdx.x;
  long long stride = (long long)gridDim.x * blockDim.x;
  for (; i < n; i += stride) d[i] = (__bf16)s[i];
}

// Wt[n, k] = W[k, n]  (one-time transpose + downconvert of weights)
__global__ void cvt_bf16_T(const float* __restrict__ W, __bf16* __restrict__ Wt,
                           int K, int N) {
  int n = blockIdx.x;
  for (int k = threadIdx.x; k < K; k += blockDim.x)
    Wt[(long long)n * K + k] = (__bf16)W[(long long)k * N + n];
}

// AGG[i,:] = dinv[i]^2 * T[i,:]   (self-loop term; also initializes AGG)
__global__ void selfloop_init(const float* __restrict__ dinv, const float* __restrict__ T,
                              float* __restrict__ AGG, int F) {
  long long i = blockIdx.x;
  float s = dinv[i] * dinv[i];
  const float* src = T + i * F;
  float*       dst = AGG + i * F;
  for (int f = threadIdx.x; f < F; f += blockDim.x) dst[f] = s * src[f];
}

// AGG[col,:] += dinv[row]*ew*dinv[col] * T[row,:]
__global__ void edge_scatter(const int* __restrict__ ei, const float* __restrict__ ew,
                             const float* __restrict__ dinv, const float* __restrict__ T,
                             float* __restrict__ AGG, int E, int F) {
  int e = blockIdx.x;
  int r = ei[e];
  int c = ei[E + e];
  float norm = dinv[r] * ew[e] * dinv[c];
  const float* src = T + (long long)r * F;
  float*       dst = AGG + (long long)c * F;
  for (int f = threadIdx.x; f < F; f += blockDim.x)
    atomicAdd(dst + f, norm * src[f]);
}

// mode 0: leaky_relu(0.01) -> bf16 ; mode 1: relu -> bf16 ; mode 2: relu -> f32
__global__ void bias_act(const float* __restrict__ AGG, const float* __restrict__ bias,
                         __bf16* __restrict__ out_bf, float* __restrict__ out_f,
                         int F, int mode) {
  long long i = blockIdx.x;
  for (int f = threadIdx.x; f < F; f += blockDim.x) {
    float v = AGG[i * F + f] + bias[f];
    if (mode == 0) v = v > 0.f ? v : 0.01f * v;
    else           v = v > 0.f ? v : 0.f;
    if (mode == 2) out_f[i * F + f] = v;
    else           out_bf[i * F + f] = (__bf16)v;
  }
}

// ---------------------------------------------------------------------------
// Host orchestration
// ---------------------------------------------------------------------------
extern "C" void kernel_launch(void* const* d_in, const int* in_sizes, int n_in,
                              void* d_out, int out_size, void* d_ws, size_t ws_size,
                              hipStream_t stream) {
  (void)n_in; (void)out_size; (void)ws_size;
  const float* x  = (const float*)d_in[0];
  const int*   ei = (const int*)d_in[1];
  const float* ew = (const float*)d_in[2];
  const float* W1 = (const float*)d_in[3];
  const float* b1 = (const float*)d_in[4];
  const float* W2 = (const float*)d_in[5];
  const float* b2 = (const float*)d_in[6];
  const float* W3 = (const float*)d_in[7];
  const float* b3 = (const float*)d_in[8];

  const int IN_F = 512, H1 = 2048, H2 = 256, OUT_F = 64;
  const int NN = in_sizes[0] / IN_F;   // 50000
  const int NE = in_sizes[2];          // 200000

  char*  base = (char*)d_ws;
  size_t off  = 0;
  auto take = [&](size_t bytes) -> void* {
    void* p = base + off;
    off += (bytes + 255) & ~(size_t)255;
    return p;
  };
  float*  dinv = (float*)take((size_t)NN * 4);
  __bf16* W1t  = (__bf16*)take((size_t)H1 * IN_F * 2);
  __bf16* W2t  = (__bf16*)take((size_t)H2 * H1 * 2);
  __bf16* W3t  = (__bf16*)take((size_t)OUT_F * H2 * 2);
  __bf16* Xb   = (__bf16*)take((size_t)NN * IN_F * 2);
  __bf16* Hb   = (__bf16*)take((size_t)NN * H1 * 2);
  float*  T    = (float*)take((size_t)NN * H1 * 4);
  float*  AGG  = (float*)take((size_t)NN * H1 * 4);

  // --- gcn_norm ---
  init_deg<<<(NN + 255) / 256, 256, 0, stream>>>(dinv, NN);
  scatter_deg<<<(NE + 255) / 256, 256, 0, stream>>>(ei, ew, dinv, NE);
  finalize_dinv<<<(NN + 255) / 256, 256, 0, stream>>>(dinv, NN);

  // --- one-time bf16 conversions ---
  cvt_bf16<<<2048, 256, 0, stream>>>(x, Xb, (long long)NN * IN_F);
  cvt_bf16_T<<<H1, 256, 0, stream>>>(W1, W1t, IN_F, H1);
  cvt_bf16_T<<<H2, 256, 0, stream>>>(W2, W2t, H1, H2);
  cvt_bf16_T<<<OUT_F, 256, 0, stream>>>(W3, W3t, H2, OUT_F);

  const int gM = (NN + GEMM_BM - 1) / GEMM_BM;
  auto layer = [&](const __bf16* Ain, const __bf16* Wt, const float* bias,
                   int K, int F, int mode) {
    dim3 grid(gM, F / GEMM_BN);
    gemm_bf16_wmma<<<grid, 256, 0, stream>>>(Ain, Wt, T, NN, F, K);
    int bt = F < 256 ? F : 256;
    selfloop_init<<<NN, bt, 0, stream>>>(dinv, T, AGG, F);
    edge_scatter<<<NE, bt, 0, stream>>>(ei, ew, dinv, T, AGG, NE, F);
    bias_act<<<NN, bt, 0, stream>>>(AGG, bias, Hb, (float*)d_out, F, mode);
  };

  layer(Xb, W1t, b1, IN_F, H1, /*leaky->bf16*/ 0);
  layer(Hb, W2t, b2, H1, H2, /*relu->bf16*/ 1);
  layer(Hb, W3t, b3, H2, OUT_F, /*relu->f32 d_out*/ 2);
}
